// GraphTransformerLayer_38285338476697
// MI455X (gfx1250) — matrix operand order repaired and checked by reading, not compiled, not run on hardware
//
#include <hip/hip_runtime.h>
#include <hip/hip_bf16.h>
#include <math.h>
#include <stdint.h>

// ---------------- problem constants (match reference) ----------------
#define NNODES 50000
#define NEDGES 400000
#define DMODEL 128
#define DFFN   256
#define NHEADS 8
#define DHEAD  16

typedef __attribute__((ext_vector_type(16))) __bf16 v16bf;
typedef __attribute__((ext_vector_type(8)))  float  v8f;

// ---------------- weight pack: f32 [K x Nc] -> bf16 WMMA B-fragment order ----
// Layout: for colTile ct (16 cols), K-block kb (32 rows):
//   out[ (ct*(K/32) + kb)*512 + lane*16 + i ]
// lane = ksel*16 + n ; element i -> k = kb*32 + ksel*16 + i ; col = ct*16 + n.
__global__ void pack_w_kernel(const float* __restrict__ W,
                              __bf16* __restrict__ out, int K, int Nc) {
  int o = blockIdx.x * blockDim.x + threadIdx.x;
  if (o >= K * Nc) return;
  const int kBlocks = K >> 5;
  int i    = o & 15;
  int lane = (o >> 4) & 31;
  int kbT  = (o >> 9) % kBlocks;
  int colT = o / (kBlocks << 9);
  int ksel = lane >> 4, n = lane & 15;
  int k   = kbT * 32 + ksel * 16 + i;
  int col = colT * 16 + n;
  out[o] = (__bf16)W[(size_t)k * Nc + col];
}

// ---------------- WMMA GEMM: wave computes a 16 x (CT*16) strip ----------------
// C[M x NC] = A[M x KT] (fp32 -> bf16 on load) @ Wp (pre-packed bf16, staged in LDS)
// FLAGS bit0: += bias[col]   bit1: += resid[row,col]
template<int KT, int CT, int FLAGS, int NC>
__global__ __launch_bounds__(256) void wmma_gemm_kernel(
    const float* __restrict__ A, const __bf16* __restrict__ Wp,
    const float* __restrict__ bias, const float* __restrict__ resid,
    float* __restrict__ C, int M)
{
  // ---- stage the whole packed weight panel (KT x NC bf16) in LDS using the
  // CDNA5 async Tensor/LDS DMA path: global_load_async_to_lds_b128 writes LDS
  // directly (no VGPR round-trip), tracked by ASYNCcnt. ----
  __shared__ __bf16 smem[KT * NC];
  {
    // flat pointer -> LDS byte offset (low 32 bits of the flat address)
    const uint32_t lds_base = (uint32_t)(uintptr_t)smem;
    const uint64_t gbase    = (uint64_t)(uintptr_t)Wp;
    constexpr int CHUNKS = (KT * NC * 2) / 16;       // 16-byte chunks
#pragma unroll
    for (int i = 0; i < CHUNKS / 256; ++i) {
      uint32_t idx      = i * 256 + threadIdx.x;
      uint32_t lds_addr = lds_base + idx * 16;
      uint64_t gaddr    = gbase + (uint64_t)idx * 16;
      asm volatile("global_load_async_to_lds_b128 %0, %1, off"
                   :: "v"(lds_addr), "v"(gaddr) : "memory");
    }
    asm volatile("s_wait_asynccnt 0x0" ::: "memory");
  }
  __syncthreads();

  const int lane   = threadIdx.x & 31;
  const int waveId = blockIdx.x * (blockDim.x >> 5) + (threadIdx.x >> 5);
  constexpr int groupsPerRow = NC / (CT * 16);
  const int rowTile  = waveId / groupsPerRow;
  const int colGroup = waveId % groupsPerRow;
  if (rowTile >= (M >> 4)) return;          // wave-uniform: EXEC all-1 at WMMA

  const int m    = lane & 15;
  const int ksel = lane >> 4;
  const int row  = rowTile * 16 + m;
  const float* __restrict__ Arow = A + (size_t)row * KT;
  const __bf16* __restrict__ SL  = smem + lane * 16;

  v8f acc[CT] = {};
#pragma unroll
  for (int kb = 0; kb < KT; kb += 32) {
    __builtin_prefetch(Arow + kb + 64, 0, 3);        // near-cache prefetch
    // A fragment: one contiguous 16-float run per lane (k = 16*ksel + i)
    float4 a0 = *(const float4*)(Arow + kb + ksel * 16);
    float4 a1 = *(const float4*)(Arow + kb + ksel * 16 + 4);
    float4 a2 = *(const float4*)(Arow + kb + ksel * 16 + 8);
    float4 a3 = *(const float4*)(Arow + kb + ksel * 16 + 12);
    v16bf a;
    a[0]=(__bf16)a0.x;  a[1]=(__bf16)a0.y;  a[2]=(__bf16)a0.z;  a[3]=(__bf16)a0.w;
    a[4]=(__bf16)a1.x;  a[5]=(__bf16)a1.y;  a[6]=(__bf16)a1.z;  a[7]=(__bf16)a1.w;
    a[8]=(__bf16)a2.x;  a[9]=(__bf16)a2.y;  a[10]=(__bf16)a2.z; a[11]=(__bf16)a2.w;
    a[12]=(__bf16)a3.x; a[13]=(__bf16)a3.y; a[14]=(__bf16)a3.z; a[15]=(__bf16)a3.w;

    // fetch all CT B-fragments (ds_load_b128 pairs overlap the WMMA chain)
    v16bf bf[CT];
#pragma unroll
    for (int ct = 0; ct < CT; ++ct)
      bf[ct] = *(const v16bf*)(SL +
          ((size_t)(colGroup * CT + ct) * (KT >> 5) + (kb >> 5)) * 512);
#pragma unroll
    for (int ct = 0; ct < CT; ++ct)
      acc[ct] = __builtin_amdgcn_wmma_f32_16x16x32_bf16(
          false, a, false, bf[ct], (short)0, acc[ct], false, false);
  }

  // C/D layout: VGPR r -> M = r + 8*ksel ; N = lane & 15
#pragma unroll
  for (int ct = 0; ct < CT; ++ct) {
    const int col = colGroup * CT * 16 + ct * 16 + m;
    const float bb = (FLAGS & 1) ? bias[col] : 0.0f;
#pragma unroll
    for (int r = 0; r < 8; ++r) {
      int    orow = rowTile * 16 + r + ksel * 8;
      size_t idx  = (size_t)orow * NC + col;
      float  val  = acc[ct][r] + bb;
      if (FLAGS & 2) val += resid[idx];
      C[idx] = val;
    }
  }
}

// ---------------- edge attention (gather/score/scatter) ----------------
__global__ __launch_bounds__(256) void edge_attn_kernel(
    const float* __restrict__ Qh, const float* __restrict__ Kh,
    const float* __restrict__ Vh, const float* __restrict__ Pe,
    const float* __restrict__ Pkr, const int* __restrict__ src,
    const int* __restrict__ dst, float* __restrict__ e_out,
    float* __restrict__ wV, float* __restrict__ z)
{
  const int edge = blockIdx.x * (blockDim.x >> 5) + (threadIdx.x >> 5);
  if (edge >= NEDGES) return;
  const int lane = threadIdx.x & 31;
  const int d    = lane * 4;
  const int s    = src[edge];
  const int t    = dst[edge];

  float4 k4 = *(const float4*)(Kh  + (size_t)s    * DMODEL + d);
  float4 q4 = *(const float4*)(Qh  + (size_t)t    * DMODEL + d);
  float4 pe = *(const float4*)(Pe  + (size_t)edge * DMODEL + d);
  float4 pk = *(const float4*)(Pkr + (size_t)edge * DMODEL + d);

  const float inv_sqrt_dh = 0.25f;   // 1/sqrt(16)
  float4 sc;
  sc.x = (k4.x * q4.x * inv_sqrt_dh + pk.x) * pe.x;
  sc.y = (k4.y * q4.y * inv_sqrt_dh + pk.y) * pe.y;
  sc.z = (k4.z * q4.z * inv_sqrt_dh + pk.z) * pe.z;
  sc.w = (k4.w * q4.w * inv_sqrt_dh + pk.w) * pe.w;
  *(float4*)(e_out + (size_t)edge * DMODEL + d) = sc;

  float part = sc.x + sc.y + sc.z + sc.w;          // per-head (4-lane) reduce
  part += __shfl_xor(part, 1, 32);
  part += __shfl_xor(part, 2, 32);
  float ssum = __expf(fminf(fmaxf(part, -5.0f), 5.0f));

  float4 v4 = *(const float4*)(Vh + (size_t)s * DMODEL + d);
  float* wvp = wV + (size_t)t * DMODEL + d;
  atomicAdd(wvp + 0, v4.x * ssum);
  atomicAdd(wvp + 1, v4.y * ssum);
  atomicAdd(wvp + 2, v4.z * ssum);
  atomicAdd(wvp + 3, v4.w * ssum);
  if ((lane & 3) == 0) atomicAdd(z + (size_t)t * NHEADS + (lane >> 2), ssum);
}

// ---------------- h_attn = wV / (z + 1e-6), in place ----------------
__global__ void attn_norm_kernel(float* __restrict__ wV,
                                 const float* __restrict__ z, int total) {
  int i = blockIdx.x * blockDim.x + threadIdx.x;
  if (i < total) {
    int node = i >> 7;
    int head = (i >> 4) & 7;
    wV[i] = wV[i] / (z[(size_t)node * NHEADS + head] + 1e-6f);
  }
}

// ---------------- LayerNorm over 128 cols, one wave per row ----------------
__global__ __launch_bounds__(256) void layernorm_kernel(
    const float* __restrict__ x, const float* __restrict__ g,
    const float* __restrict__ b, float* __restrict__ y, int M)
{
  const int row = blockIdx.x * (blockDim.x >> 5) + (threadIdx.x >> 5);
  if (row >= M) return;
  const int lane = threadIdx.x & 31;
  const int c    = lane * 4;
  float4 v = *(const float4*)(x + (size_t)row * DMODEL + c);

  float s = v.x + v.y + v.z + v.w;
#pragma unroll
  for (int o = 16; o >= 1; o >>= 1) s += __shfl_xor(s, o, 32);
  const float mu = s * (1.0f / 128.0f);

  float dx = v.x - mu, dy = v.y - mu, dz = v.z - mu, dw = v.w - mu;
  float q = dx * dx + dy * dy + dz * dz + dw * dw;
#pragma unroll
  for (int o = 16; o >= 1; o >>= 1) q += __shfl_xor(q, o, 32);
  const float inv = rsqrtf(q * (1.0f / 128.0f) + 1e-5f);

  float4 gv = *(const float4*)(g + c);
  float4 bv = *(const float4*)(b + c);
  float4 out;
  out.x = dx * inv * gv.x + bv.x;
  out.y = dy * inv * gv.y + bv.y;
  out.z = dz * inv * gv.z + bv.z;
  out.w = dw * inv * gv.w + bv.w;
  *(float4*)(y + (size_t)row * DMODEL + c) = out;
}

// ---------------- y = resid + gelu_exact(x) ----------------
__global__ void gelu_add_kernel(const float* __restrict__ x,
                                const float* __restrict__ resid,
                                float* __restrict__ y, int n) {
  int i = blockIdx.x * blockDim.x + threadIdx.x;
  if (i < n) {
    float v  = x[i];
    float gl = 0.5f * v * (1.0f + erff(v * 0.70710678118654752f));
    y[i] = resid[i] + gl;
  }
}

// ============================================================================
extern "C" void kernel_launch(void* const* d_in, const int* in_sizes, int n_in,
                              void* d_out, int out_size, void* d_ws, size_t ws_size,
                              hipStream_t stream) {
  (void)in_sizes; (void)n_in; (void)out_size; (void)ws_size;

  // -------- inputs (setup_inputs dict order, params dict flattened) --------
  const float* h    = (const float*)d_in[0];
  const float* e    = (const float*)d_in[1];
  const float* kr   = (const float*)d_in[2];
  const int*   src  = (const int*)d_in[3];
  const int*   dst  = (const int*)d_in[4];
  const float* Wq   = (const float*)d_in[5];
  const float* Wk   = (const float*)d_in[6];
  const float* Wv   = (const float*)d_in[7];
  const float* We   = (const float*)d_in[8];
  const float* Wkr  = (const float*)d_in[9];
  const float* Wo_h = (const float*)d_in[10];
  const float* bo_h = (const float*)d_in[11];
  const float* Wo_e = (const float*)d_in[12];
  const float* bo_e = (const float*)d_in[13];
  const float* ln1h_g = (const float*)d_in[14];
  const float* ln1h_b = (const float*)d_in[15];
  const float* ln1e_g = (const float*)d_in[16];
  const float* ln1e_b = (const float*)d_in[17];
  const float* ln2h_g = (const float*)d_in[18];
  const float* ln2h_b = (const float*)d_in[19];
  const float* ln2e_g = (const float*)d_in[20];
  const float* ln2e_b = (const float*)d_in[21];
  const float* W1h  = (const float*)d_in[22];
  const float* b1h  = (const float*)d_in[23];
  const float* W2h  = (const float*)d_in[24];
  const float* b2h  = (const float*)d_in[25];
  const float* W1e  = (const float*)d_in[26];
  const float* b1e  = (const float*)d_in[27];
  const float* W2e  = (const float*)d_in[28];
  const float* b2e  = (const float*)d_in[29];

  // -------- workspace layout --------
  const size_t DD = (size_t)DMODEL * DMODEL;   // 16384
  const size_t DF = (size_t)DMODEL * DFFN;     // 32768
  __bf16* WB = (__bf16*)d_ws;
  __bf16 *pWq  = WB,           *pWk  = pWq  + DD, *pWv  = pWk  + DD,
         *pWe  = pWv  + DD,    *pWkr = pWe  + DD, *pWoh = pWkr + DD,
         *pWoe = pWoh + DD,    *pW1h = pWoe + DD, *pW2h = pW1h + DF,
         *pW1e = pW2h + DF,    *pW2e = pW1e + DF;
  // bf16 region ends at 245760 elems; float region starts at 131072 floats
  float* F = (float*)d_ws + 131072;
  const size_t N128 = (size_t)NNODES * DMODEL;   // 6.4M
  const size_t E128 = (size_t)NEDGES * DMODEL;   // 51.2M
  float* Qh  = F;               float* Kh  = Qh  + N128;
  float* Vh  = Kh + N128;       float* wV  = Vh  + N128;   // later: h_attn in place
  float* H2  = wV + N128;       float* T1H = H2  + N128;   // N x 256
  float* T2H = T1H + 2 * N128;  float* Zb  = T2H + N128;   // N x 8
  float* P1  = Zb + (size_t)NNODES * NHEADS;                // proj_e   (later T1E lo)
  float* P2  = P1 + E128;                                   // proj_kr  (later T1E hi)
  float* EO  = P2 + E128;                                   // e_out    (later T2E)
  float* E2  = EO + E128;                                   // e2
  float* T1E = P1;                                          // E x 256 reuse

  float* out_h = (float*)d_out;
  float* out_e = out_h + N128;

  const int TPB = 256;
  auto cdiv = [](long long a, long long b) { return (int)((a + b - 1) / b); };
  auto pack = [&](const float* s, __bf16* d, int K, int Nc) {
    pack_w_kernel<<<cdiv((long long)K * Nc, TPB), TPB, 0, stream>>>(s, d, K, Nc);
  };
  auto blocksFor = [&](int M, int Nc, int CT) {
    long long waves = (long long)(M >> 4) * (Nc / (CT * 16));
    return cdiv(waves, 8);
  };

  // -------- 0: zero the scatter accumulators --------
  hipMemsetAsync(wV, 0, N128 * sizeof(float), stream);
  hipMemsetAsync(Zb, 0, (size_t)NNODES * NHEADS * sizeof(float), stream);

  // -------- 1: weights -> packed bf16 fragments --------
  pack(Wq,  pWq,  DMODEL, DMODEL);  pack(Wk,  pWk,  DMODEL, DMODEL);
  pack(Wv,  pWv,  DMODEL, DMODEL);  pack(We,  pWe,  DMODEL, DMODEL);
  pack(Wkr, pWkr, DMODEL, DMODEL);  pack(Wo_h, pWoh, DMODEL, DMODEL);
  pack(Wo_e, pWoe, DMODEL, DMODEL);
  pack(W1h, pW1h, DMODEL, DFFN);    pack(W2h, pW2h, DFFN, DMODEL);
  pack(W1e, pW1e, DMODEL, DFFN);    pack(W2e, pW2e, DFFN, DMODEL);

  // -------- 2: projections (WMMA GEMMs, K=128, Nc=128) --------
  wmma_gemm_kernel<128,8,0,128><<<blocksFor(NNODES,128,8), TPB, 0, stream>>>(h,  pWq,  nullptr, nullptr, Qh, NNODES);
  wmma_gemm_kernel<128,8,0,128><<<blocksFor(NNODES,128,8), TPB, 0, stream>>>(h,  pWk,  nullptr, nullptr, Kh, NNODES);
  wmma_gemm_kernel<128,8,0,128><<<blocksFor(NNODES,128,8), TPB, 0, stream>>>(h,  pWv,  nullptr, nullptr, Vh, NNODES);
  wmma_gemm_kernel<128,8,0,128><<<blocksFor(NEDGES,128,8), TPB, 0, stream>>>(e,  pWe,  nullptr, nullptr, P1, NEDGES);
  wmma_gemm_kernel<128,8,0,128><<<blocksFor(NEDGES,128,8), TPB, 0, stream>>>(kr, pWkr, nullptr, nullptr, P2, NEDGES);

  // -------- 3: edge attention scores + scatter --------
  edge_attn_kernel<<<cdiv((long long)NEDGES, 8), TPB, 0, stream>>>(
      Qh, Kh, Vh, P1, P2, src, dst, EO, wV, Zb);

  // -------- 4: h_attn = wV / (z + 1e-6) --------
  attn_norm_kernel<<<cdiv((long long)N128, TPB), TPB, 0, stream>>>(wV, Zb, (int)N128);

  // -------- 5: output projections + residual, then LN1 --------
  wmma_gemm_kernel<128,8,3,128><<<blocksFor(NNODES,128,8), TPB, 0, stream>>>(wV, pWoh, bo_h, h, H2, NNODES);
  layernorm_kernel<<<cdiv((long long)NNODES, 8), TPB, 0, stream>>>(H2, ln1h_g, ln1h_b, H2, NNODES);
  wmma_gemm_kernel<128,8,3,128><<<blocksFor(NEDGES,128,8), TPB, 0, stream>>>(EO, pWoe, bo_e, e, E2, NEDGES);
  layernorm_kernel<<<cdiv((long long)NEDGES, 8), TPB, 0, stream>>>(E2, ln1e_g, ln1e_b, E2, NEDGES);

  // -------- 6: node FFN: gelu((h2@W1h+b1h)@W2h+b2h), residual, LN2 --------
  wmma_gemm_kernel<128,8,1,256><<<blocksFor(NNODES,256,8), TPB, 0, stream>>>(H2,  pW1h, b1h, nullptr, T1H, NNODES);
  wmma_gemm_kernel<256,8,1,128><<<blocksFor(NNODES,128,8), TPB, 0, stream>>>(T1H, pW2h, b2h, nullptr, T2H, NNODES);
  gelu_add_kernel<<<cdiv((long long)N128, TPB), TPB, 0, stream>>>(T2H, H2, T2H, (int)N128);
  layernorm_kernel<<<cdiv((long long)NNODES, 8), TPB, 0, stream>>>(T2H, ln2h_g, ln2h_b, out_h, NNODES);

  // -------- 7: edge FFN (reusing dead proj/e_out buffers) --------
  wmma_gemm_kernel<128,8,1,256><<<blocksFor(NEDGES,256,8), TPB, 0, stream>>>(E2,  pW1e, b1e, nullptr, T1E, NEDGES);
  wmma_gemm_kernel<256,8,1,128><<<blocksFor(NEDGES,128,8), TPB, 0, stream>>>(T1E, pW2e, b2e, nullptr, EO,  NEDGES);
  gelu_add_kernel<<<cdiv((long long)E128, TPB), TPB, 0, stream>>>(EO, E2, EO, (int)E128);
  layernorm_kernel<<<cdiv((long long)NEDGES, 8), TPB, 0, stream>>>(EO, ln2e_g, ln2e_b, out_e, NEDGES);
}
